// Triton_MHA_77687368450156
// MI455X (gfx1250) — matrix operand order repaired and checked by reading, not compiled, not run on hardware
//
#include <hip/hip_runtime.h>

typedef __attribute__((ext_vector_type(16))) _Float16 v16h;
typedef __attribute__((ext_vector_type(8)))  _Float16 v8h;
typedef __attribute__((ext_vector_type(2)))  _Float16 v2h;
typedef __attribute__((ext_vector_type(8)))  float    v8f;
typedef __attribute__((ext_vector_type(4)))  float    v4f;

#define B_  8
#define L_  1000
#define D_  1024
#define H_  16
#define DH  64
#define LP  1024   // padded sequence length

// ---------------------------------------------------------------------------
// Prep: Qh[b,h,i,d] = f16(query[b,i,h*64+d]) (zero-padded i>=L), and the
// d-major transpose Vt[b,h,d,i] of the same data (V == Q for self-attention).
// ---------------------------------------------------------------------------
__global__ __launch_bounds__(256)
void prep_kernel(const float* __restrict__ q,
                 _Float16* __restrict__ Qh, _Float16* __restrict__ Vt) {
  int idx = blockIdx.x * blockDim.x + threadIdx.x;   // B*H*LP*DH = 2^23
  int d = idx & (DH - 1);
  int i = (idx >> 6) & (LP - 1);
  int h = (idx >> 16) & (H_ - 1);
  int b = idx >> 20;
  float val = 0.0f;
  if (i < L_) val = q[((size_t)(b * L_ + i)) * D_ + h * DH + d];
  _Float16 hv = (_Float16)val;
  Qh[idx] = hv;                                          // ((b*H+h)*LP+i)*DH+d
  Vt[((size_t)(b * H_ + h) * DH + d) * LP + i] = hv;     // ((b*H+h)*DH+d)*LP+i
}

// mask -> additive bias: 0.0f where attendable, -1e9 where blocked/padded
__global__ __launch_bounds__(256)
void bias_kernel(const int* __restrict__ mask, float* __restrict__ bias) {
  int idx = blockIdx.x * blockDim.x + threadIdx.x;   // B*LP = 8192
  int i = idx & (LP - 1);
  int b = idx >> 10;
  int allowed = (i < L_) ? mask[b * L_ + i] : 0;
  bias[idx] = allowed ? 0.0f : -1.0e9f;
}

// ---------------------------------------------------------------------------
// S^T tile (16 keys x 16 queries) = K_tile * Q^T over dh=64, with the mask
// bias pre-loaded as the WMMA C accumulator (bias is per-key == per-VGPR,
// identical across lanes, so it drops straight into the C layout).
// ---------------------------------------------------------------------------
__device__ __forceinline__
v8f st_tile(const _Float16* krow, const float* bp,
            const v16h& bq0, const v16h& bq1, int hi) {
  v8f c;
  ((v4f*)&c)[0] = *(const v4f*)(bp);
  ((v4f*)&c)[1] = *(const v4f*)(bp + 4);
  v16h ak;
  ((v8h*)&ak)[0] = *(const v8h*)(krow + hi * 16);
  ((v8h*)&ak)[1] = *(const v8h*)(krow + hi * 16 + 8);
  c = __builtin_amdgcn_wmma_f32_16x16x32_f16(false, ak, false, bq0, (short)0, c, false, false);
  ((v8h*)&ak)[0] = *(const v8h*)(krow + 32 + hi * 16);
  ((v8h*)&ak)[1] = *(const v8h*)(krow + 32 + hi * 16 + 8);
  c = __builtin_amdgcn_wmma_f32_16x16x32_f16(false, ak, false, bq1, (short)0, c, false, false);
  return c;
}

// ---------------------------------------------------------------------------
// One 32-key chunk of P^T: exp, pack to f16 pairs, half-swap exchange to the
// B layout (4 shuffles), then O^T += V^T * P^T and l += (1 * P^T)[0].
// ---------------------------------------------------------------------------
__device__ __forceinline__
void pv_chunk(const v8f& sa, const v8f& sb, float mn, int hi,
              const _Float16* vbase, const v16h& ones,
              v8f& o0, v8f& o1, v8f& o2, v8f& o3, float& lacc) {
  unsigned pa[4], pb[4];
#pragma unroll
  for (int j = 0; j < 4; ++j) {
    pa[j] = __builtin_bit_cast(unsigned,
        __builtin_amdgcn_cvt_pkrtz(__expf(sa[2*j] - mn), __expf(sa[2*j+1] - mn)));
    pb[j] = __builtin_bit_cast(unsigned,
        __builtin_amdgcn_cvt_pkrtz(__expf(sb[2*j] - mn), __expf(sb[2*j+1] - mn)));
  }
  // Contiguous k order used here (matches V^T A): lanes hi=0 hold keys 0..15, hi=1 keys 16..31.
  // Own registers hold keys {0..7|8..15} (sa) and {16..23|24..31} (sb) by hi;
  // exchange the half each partner needs with one xor-16 shuffle per dword.
  v16h bp;
#pragma unroll
  for (int j = 0; j < 4; ++j) {
    unsigned snd = hi ? pa[j] : pb[j];
    unsigned x   = __shfl_xor(snd, 16);
    unsigned lo  = hi ? x : pa[j];
    unsigned hw  = hi ? pb[j] : x;
    ((v2h*)&bp)[j]     = __builtin_bit_cast(v2h, lo);
    ((v2h*)&bp)[j + 4] = __builtin_bit_cast(v2h, hw);
  }

  v8f ls = {};
  ls = __builtin_amdgcn_wmma_f32_16x16x32_f16(false, ones, false, bp, (short)0, ls, false, false);
  lacc += ls[0];

  v16h av;
#define PV_TILE(OT, TOFF)                                                     \
  ((v8h*)&av)[0] = *(const v8h*)(vbase + (TOFF) + hi * 16);                   \
  ((v8h*)&av)[1] = *(const v8h*)(vbase + (TOFF) + hi * 16 + 8);               \
  OT = __builtin_amdgcn_wmma_f32_16x16x32_f16(false, av, false, bp, (short)0, OT, false, false);
  PV_TILE(o0, 0 * 16 * LP)
  PV_TILE(o1, 1 * 16 * LP)
  PV_TILE(o2, 2 * 16 * LP)
  PV_TILE(o3, 3 * 16 * LP)
#undef PV_TILE
}

// ---------------------------------------------------------------------------
// Transposed flash attention: one wave32 per 16-query tile, 64-key steps.
// ---------------------------------------------------------------------------
__global__ __launch_bounds__(128)
void attn_kernel(const _Float16* __restrict__ Qh, const _Float16* __restrict__ Vt,
                 const float* __restrict__ bias, float* __restrict__ out) {
  __shared__ __align__(16) float sO[4][16 * 64];   // per-wave epilogue transpose

  const int lane = threadIdx.x & 31;
  const int wave = threadIdx.x >> 5;
  const int n    = lane & 15;     // tile column (query for S^T/O^T)
  const int hi   = lane >> 4;

  const int bh  = blockIdx.x >> 4;            // b*H + h
  const int grp = blockIdx.x & 15;
  const int b   = bh >> 4;
  const int h   = bh & 15;
  const int q0  = (grp * 4 + wave) * 16;

  const _Float16* __restrict__ Qbh = Qh + (size_t)bh * LP * DH;  // [i][d]
  const _Float16* __restrict__ Vbh = Vt + (size_t)bh * DH * LP;  // [d][i]
  const float* __restrict__ bb = bias + b * LP;
  float* so = sO[wave];

  // --- Q^T as the (loop-invariant) B operand, pre-scaled by 0.125 (exact) --
  const _Float16* qrow = Qbh + (size_t)(q0 + n) * DH;
  v16h bq0, bq1;
  ((v8h*)&bq0)[0] = *(const v8h*)(qrow + hi * 16);
  ((v8h*)&bq0)[1] = *(const v8h*)(qrow + hi * 16 + 8);
  ((v8h*)&bq1)[0] = *(const v8h*)(qrow + 32 + hi * 16);
  ((v8h*)&bq1)[1] = *(const v8h*)(qrow + 32 + hi * 16 + 8);
#pragma unroll
  for (int j = 0; j < 16; ++j) {
    bq0[j] *= (_Float16)0.125f;
    bq1[j] *= (_Float16)0.125f;
  }

  v16h ones;
#pragma unroll
  for (int j = 0; j < 16; ++j) ones[j] = (_Float16)1.0f;

  float mrun = -3.0e38f, lrun = 0.0f;
  v8f o0 = {}, o1 = {}, o2 = {}, o3 = {};

  for (int kt = 0; kt < LP / 64; ++kt) {
    const int kb = kt * 64;
    const _Float16* krow  = Qbh + (size_t)(kb + n) * DH;
    const _Float16* vbase = Vbh + (size_t)n * LP + kb;

    __builtin_prefetch(krow + 64 * DH, 0, 3);
    __builtin_prefetch(vbase + 64, 0, 3);

    // ---- S^T tiles (keys in rows), mask bias folded into C ---------------
    v8f s0 = st_tile(krow,           bb + kb +      8 * hi, bq0, bq1, hi);
    v8f s1 = st_tile(krow + 16 * DH, bb + kb + 16 + 8 * hi, bq0, bq1, hi);
    v8f s2 = st_tile(krow + 32 * DH, bb + kb + 32 + 8 * hi, bq0, bq1, hi);
    v8f s3 = st_tile(krow + 48 * DH, bb + kb + 48 + 8 * hi, bq0, bq1, hi);

    // ---- per-lane softmax state: 31 local max + one xor-16 shuffle -------
    float t = fmaxf(fmaxf(s0[0], s1[0]), fmaxf(s2[0], s3[0]));
#pragma unroll
    for (int r = 1; r < 8; ++r)
      t = fmaxf(t, fmaxf(fmaxf(s0[r], s1[r]), fmaxf(s2[r], s3[r])));
    t = fmaxf(t, __shfl_xor(t, 16));
    float mn   = fmaxf(mrun, t);
    float corr = __expf(mrun - mn);
    mrun = mn;
    lrun *= corr;
#pragma unroll
    for (int r = 0; r < 8; ++r) {
      o0[r] *= corr; o1[r] *= corr; o2[r] *= corr; o3[r] *= corr;
    }

    // ---- P^T -> f16, half-swap to B layout, O^T += V^T P^T, l += 1*P^T ---
    pv_chunk(s0, s1, mn, hi, vbase,      ones, o0, o1, o2, o3, lrun);
    pv_chunk(s2, s3, mn, hi, vbase + 32, ones, o0, o1, o2, o3, lrun);
  }

  // --- epilogue: normalize, LDS transpose, coalesced stores ----------------
  float inv = 1.0f / lrun;
#pragma unroll
  for (int r = 0; r < 8; ++r) {
    int dr = r + 8 * hi;
    so[(0 * 16 + dr) * 16 + n] = o0[r] * inv;   // sO[d][query]
    so[(1 * 16 + dr) * 16 + n] = o1[r] * inv;
    so[(2 * 16 + dr) * 16 + n] = o2[r] * inv;
    so[(3 * 16 + dr) * 16 + n] = o3[r] * inv;
  }
  asm volatile("s_wait_dscnt 0" ::: "memory");

  const int qq = lane >> 1;            // query handled by this lane
  const int dd = (lane & 1) * 32;      // d-half handled by this lane
  const int qi = q0 + qq;
  if (qi < L_) {
    float* op = out + ((size_t)(b * L_ + qi)) * D_ + h * DH + dd;
#pragma unroll
    for (int j = 0; j < 8; ++j) {
      v4f w;
      w[0] = so[(dd + 4 * j + 0) * 16 + qq];
      w[1] = so[(dd + 4 * j + 1) * 16 + qq];
      w[2] = so[(dd + 4 * j + 2) * 16 + qq];
      w[3] = so[(dd + 4 * j + 3) * 16 + qq];
      *(v4f*)(op + 4 * j) = w;
    }
  }
}

// ---------------------------------------------------------------------------
extern "C" void kernel_launch(void* const* d_in, const int* in_sizes, int n_in,
                              void* d_out, int out_size, void* d_ws, size_t ws_size,
                              hipStream_t stream) {
  (void)in_sizes; (void)n_in; (void)out_size; (void)ws_size;
  const float* query = (const float*)d_in[0];
  const int*   mask  = (const int*)d_in[1];
  float*       out   = (float*)d_out;

  _Float16* Qh   = (_Float16*)d_ws;                          // 16 MiB
  _Float16* Vt   = Qh + (size_t)B_ * H_ * LP * DH;           // 16 MiB
  float*    bias = (float*)(Vt + (size_t)B_ * H_ * LP * DH); // 32 KiB

  const int total = B_ * H_ * LP * DH;                       // 2^23
  prep_kernel<<<total / 256, 256, 0, stream>>>(query, Qh, Vt);
  bias_kernel<<<(B_ * LP) / 256, 256, 0, stream>>>(mask, bias);

  // 2048 blocks * 4 waves = 8192 waves, one 16-query tile each
  attn_kernel<<<B_ * H_ * 16, 128, 0, stream>>>(Qh, Vt, bias, out);
}
